// MoCo_83305185673832
// MI455X (gfx1250) — matrix operand order repaired and checked by reading, not compiled, not run on hardware
//
#include <hip/hip_runtime.h>
#include <hip/hip_bf16.h>

// ---------------------------------------------------------------------------
// MI455X (gfx1250) fused graph-transformer layer.
// All GEMMs run on v_wmma_f32_16x16x32_bf16 (bf16 A/B, fp32 accumulate);
// softmax / elementwise math stays fp32. Edge tensor Y is never materialized
// in HBM: per (b, q-tile, k-split) workgroup we fuse E1/E2 GEMMs, the Y
// combine, online softmax, Z = Y@Weo, and the Wef contraction, keeping
// everything in 129KB of LDS (gfx1250 WGP has 320KB).
// ---------------------------------------------------------------------------

typedef __bf16 bf16_t;
typedef __attribute__((ext_vector_type(16))) __bf16 v16bf;
typedef __attribute__((ext_vector_type(8)))  float  v8f;

#define BS 2
#define NN 512
#define DX 128
#define DE 32
#define SPLIT 4          // k-range split -> 64*4 = 256 workgroups for the big kernel
#define KT_PER_SPLIT 8   // 8 k-tiles of 16 -> 128 k per split

__device__ __forceinline__ float neg_inf() { return -__builtin_inff(); }

__device__ __forceinline__ bf16_t f2bf(float f) {
  union { float f; unsigned u; } in; in.f = f;
  unsigned u = in.u;
  unsigned r = u + 0x7FFFu + ((u >> 16) & 1u);   // round-to-nearest-even
  union { unsigned short s; bf16_t b; } out; out.s = (unsigned short)(r >> 16);
  return out.b;
}
__device__ __forceinline__ float bf2f(bf16_t b) {
  union { bf16_t b; unsigned short s; } in; in.b = b;
  union { unsigned u; float f; } out; out.u = ((unsigned)in.s) << 16;
  return out.f;
}

// A fragment (16x32 bf16, row-major source): lane holds M = lane%16,
// K(e) = 16*(e/8) + (e%8) + 8*(lane/16)  (ISA 7.12.2 16-bit A layout)
__device__ __forceinline__ v16bf load_a_frag(const bf16_t* base, int ld, int k0, int lane) {
  int m = lane & 15, half = lane >> 4;
  const bf16_t* p = base + m * ld + k0 + 8 * half;
  union { uint4 u[2]; v16bf v; } t;
  t.u[0] = *(const uint4*)(p);
  t.u[1] = *(const uint4*)(p + 16);
  return t.v;
}

// B fragment (32x16 bf16) loaded from a PRE-TRANSPOSED weight (N x K row
// major): lane holds N = row n, K(e) = 16*(e/8) + (e%8) + 8*(lane/16).
__device__ __forceinline__ v16bf load_b_frag(const bf16_t* bt, int ld, int n, int k0, int lane) {
  const bf16_t* p = bt + (size_t)n * ld + k0 + 8 * (lane >> 4);
  union { uint4 u[2]; v16bf v; } t;
  t.u[0] = *(const uint4*)(p);
  t.u[1] = *(const uint4*)(p + 16);
  return t.v;
}

__device__ __forceinline__ v8f wmma_bf16(v16bf a, v16bf b, v8f c) {
  return __builtin_amdgcn_wmma_f32_16x16x32_bf16(false, a, false, b, (short)0, c, false, false);
}

__device__ __forceinline__ v8f vzero8() {
  v8f c;
  #pragma unroll
  for (int i = 0; i < 8; ++i) c[i] = 0.f;
  return c;
}

// ---------------------------------------------------------------------------
// prep: f32 -> bf16 convert / transpose-convert
// ---------------------------------------------------------------------------
__global__ void cvt_bf16(const float* __restrict__ src, bf16_t* __restrict__ dst, int n) {
  for (int i = blockIdx.x * blockDim.x + threadIdx.x; i < n; i += gridDim.x * blockDim.x)
    dst[i] = f2bf(src[i]);
}
__global__ void transpose_bf16(const float* __restrict__ src, bf16_t* __restrict__ dst,
                               int rows, int cols) {
  int total = rows * cols;
  for (int i = blockIdx.x * blockDim.x + threadIdx.x; i < total; i += gridDim.x * blockDim.x) {
    int r = i / cols, c = i - r * cols;
    dst[(size_t)c * rows + r] = f2bf(src[i]);
  }
}

// ---------------------------------------------------------------------------
// QKV:  [1024x128] bf16  @  WT[128x128] bf16  ->  f32 (+bias, *mask)
// grid (64 M-tiles, 3 weights), 128 threads (4 waves), wave owns 2 N-tiles.
// ---------------------------------------------------------------------------
__global__ __launch_bounds__(128) void qkv_gemm(
    const bf16_t* __restrict__ Xbf,
    const bf16_t* __restrict__ WqT, const bf16_t* __restrict__ WkT, const bf16_t* __restrict__ WvT,
    const float* __restrict__ bq, const float* __restrict__ bk, const float* __restrict__ bv,
    const float* __restrict__ node_mask,
    float* __restrict__ Qf, float* __restrict__ Kf, float* __restrict__ Vf) {
  int which = blockIdx.y;
  const bf16_t* WT  = which == 0 ? WqT : which == 1 ? WkT : WvT;
  const float*  bia = which == 0 ? bq  : which == 1 ? bk  : bv;
  float*        out = which == 0 ? Qf  : which == 1 ? Kf  : Vf;
  int lane = threadIdx.x & 31, wave = threadIdx.x >> 5;
  int row0 = blockIdx.x * 16;
  #pragma unroll
  for (int t = 0; t < 2; ++t) {
    int n = wave * 2 + t;
    v8f c = vzero8();
    #pragma unroll
    for (int ks = 0; ks < 4; ++ks) {
      v16bf a = load_a_frag(Xbf + row0 * DX, DX, ks * 32, lane);
      v16bf b = load_b_frag(WT, DX, n * 16 + (lane & 15), ks * 32, lane);
      c = wmma_bf16(a, b, c);
    }
    int d = n * 16 + (lane & 15);
    #pragma unroll
    for (int r = 0; r < 8; ++r) {
      int row = row0 + r + 8 * (lane >> 4);
      out[row * DX + d] = (c[r] + bia[d]) * node_mask[row];
    }
  }
}

// ---------------------------------------------------------------------------
// Fused edge/attention kernel. grid(64 = b*32+qtile, SPLIT), 256 threads.
// ---------------------------------------------------------------------------
__global__ __launch_bounds__(256) void fused_attn(
    const float* __restrict__ Ef, const float* __restrict__ node_mask,
    const float* __restrict__ Qf, const float* __restrict__ Kf, const float* __restrict__ Vf,
    const bf16_t* __restrict__ WemT, const bf16_t* __restrict__ WeaT,
    const bf16_t* __restrict__ WeoT, const bf16_t* __restrict__ WefT,
    const float* __restrict__ bem, const float* __restrict__ bea, const float* __restrict__ beo,
    float* __restrict__ partM, float* __restrict__ partS, float* __restrict__ partA,
    float* __restrict__ partE) {
  __shared__ __align__(16) float  Qs[16 * DX];        // 8KB
  __shared__ __align__(16) float  Ksh[16 * DX];       // 8KB
  __shared__ __align__(16) float  Vsh[16 * DX];       // 8KB
  __shared__ __align__(16) bf16_t Es[16 * 16 * DE];   // 16KB  [q][k][de]
  __shared__ __align__(16) bf16_t Ybf[128 * DX];      // 32KB  half tile: 8q x 16k x 128d
  __shared__ __align__(16) bf16_t Zs[256 * DE];       // 16KB  [q*16+k][de]
  __shared__ __align__(16) bf16_t WemTs[DX * DE];     // 8KB
  __shared__ __align__(16) bf16_t WeaTs[DX * DE];     // 8KB
  __shared__ __align__(16) bf16_t WeoTs[DE * DX];     // 8KB
  __shared__ float bemS[DX], beaS[DX], beoS[DE];
  __shared__ float maskQs[16], maskKs[16];
  __shared__ float newEpart[8 * 16 * DE];             // 16KB per-wave Wef partials

  const int tid = threadIdx.x, lane = tid & 31, wave = tid >> 5;
  const int wg = blockIdx.x, split = blockIdx.y;
  const int b = wg >> 5, qt = wg & 31, q0 = qt * 16;

  // stage weights / biases / Q-tile once
  for (int i = tid; i < DX * DE; i += 256) { WemTs[i] = WemT[i]; WeaTs[i] = WeaT[i]; }
  for (int i = tid; i < DE * DX; i += 256) WeoTs[i] = WeoT[i];
  for (int i = tid; i < 2048; i += 256)
    Qs[i] = Qf[(b * NN + q0 + (i >> 7)) * DX + (i & 127)];
  for (int i = tid; i < DX; i += 256) { bemS[i] = bem[i]; beaS[i] = bea[i]; }
  if (tid < DE) beoS[tid] = beo[tid];
  if (tid < 16) maskQs[tid] = node_mask[b * NN + q0 + tid];

  // per-thread online-softmax state: 8 (q,d) pairs, p = j*256 + tid
  float sm[8], ss[8], sa[8];
  #pragma unroll
  for (int j = 0; j < 8; ++j) { sm[j] = neg_inf(); ss[j] = 0.f; sa[j] = 0.f; }
  v8f cW[2]; cW[0] = vzero8(); cW[1] = vzero8();     // Wef accumulators (nt=0,1)
  __syncthreads();

  const int ktEnd = split * KT_PER_SPLIT + KT_PER_SPLIT;
  for (int kt = split * KT_PER_SPLIT; kt < ktEnd; ++kt) {
    const int k0 = kt * 16;
    // prefetch next E tile into L2 (global_prefetch_b8)
    if (kt + 1 < ktEnd && tid < 64) {
      const float* pn = Ef + ((size_t)(b * NN + q0 + (tid >> 2)) * NN + (kt + 1) * 16) * DE
                        + (tid & 3) * 128;
      __builtin_prefetch(pn, 0, 1);
    }
    // stage K, V, E, k-mask
    for (int i = tid; i < 2048; i += 256) {
      int k = i >> 7, d = i & 127;
      Ksh[i] = Kf[(b * NN + k0 + k) * DX + d];
      Vsh[i] = Vf[(b * NN + k0 + k) * DX + d];
    }
    for (int i = tid; i < 8192; i += 256) {
      int q = i >> 9, k = (i >> 5) & 15, e = i & 31;
      Es[i] = f2bf(Ef[((size_t)(b * NN + q0 + q) * NN + k0 + k) * DE + e]);
    }
    if (tid < 16) maskKs[tid] = node_mask[b * NN + k0 + tid];
    __syncthreads();

    #pragma unroll
    for (int h = 0; h < 2; ++h) {
      // ---- phase A: wave w owns q = h*8 + w; E1/E2 WMMA + Y combine -> Ybf
      const int q = h * 8 + wave;
      const float mq = maskQs[q];
      #pragma unroll
      for (int n = 0; n < 8; ++n) {
        v16bf a  = load_a_frag(Es + q * (16 * DE), DE, 0, lane);
        v16bf b1 = load_b_frag(WemTs, DE, n * 16 + (lane & 15), 0, lane);
        v16bf b2 = load_b_frag(WeaTs, DE, n * 16 + (lane & 15), 0, lane);
        v8f c1 = wmma_bf16(a, b1, vzero8());
        v8f c2 = wmma_bf16(a, b2, vzero8());
        const int d = n * 16 + (lane & 15);
        const float qv = Qs[q * DX + d];
        #pragma unroll
        for (int r = 0; r < 8; ++r) {
          int kl = r + 8 * (lane >> 4);
          float mm = mq * maskKs[kl];
          float e1 = (c1[r] + bemS[d]) * mm;
          float e2 = (c2[r] + beaS[d]) * mm;
          float y  = qv * Ksh[kl * DX + d] * 0.25f;   // 1/sqrt(df=16)
          y = y * (e1 + 1.0f) + e2;
          Ybf[((q & 7) * 16 + kl) * DX + d] = f2bf(y);
        }
      }
      __syncthreads();
      // ---- phase B1: Z = Y @ Weo (+beo, *mask) -> Zs  (wave w owns its rows)
      #pragma unroll
      for (int nt = 0; nt < 2; ++nt) {
        v8f c = vzero8();
        #pragma unroll
        for (int ks = 0; ks < 4; ++ks) {
          v16bf a = load_a_frag(Ybf + wave * 16 * DX, DX, ks * 32, lane);
          v16bf bb = load_b_frag(WeoTs, DX, nt * 16 + (lane & 15), ks * 32, lane);
          c = wmma_bf16(a, bb, c);
        }
        const int e = nt * 16 + (lane & 15);
        #pragma unroll
        for (int r = 0; r < 8; ++r) {
          int kl = r + 8 * (lane >> 4);
          float z = (c[r] + beoS[e]) * mq * maskKs[kl];
          Zs[(q * 16 + kl) * DE + e] = f2bf(z);
        }
      }
      // ---- phase B2: online softmax over this half's 16 k values
      #pragma unroll
      for (int jj = 0; jj < 4; ++jj) {
        const int j = h * 4 + jj;
        const int p = j * 256 + tid;
        const int qq = p >> 7, d = p & 127;
        const int rb = (qq & 7) * 16;
        float yv[16]; float tm = neg_inf();
        #pragma unroll
        for (int k = 0; k < 16; ++k) {
          yv[k] = bf2f(Ybf[(rb + k) * DX + d]);
          if (maskKs[k] > 0.f && yv[k] > tm) tm = yv[k];
        }
        if (tm > neg_inf()) {
          float mnew = fmaxf(sm[j], tm);
          float sc = (sm[j] > neg_inf()) ? __expf(sm[j] - mnew) : 0.f;
          float s = ss[j] * sc, acc = sa[j] * sc;
          #pragma unroll
          for (int k = 0; k < 16; ++k) {
            if (maskKs[k] > 0.f) {
              float ee = __expf(yv[k] - mnew);
              s += ee; acc += ee * Vsh[k * DX + d];
            }
          }
          sm[j] = mnew; ss[j] = s; sa[j] = acc;
        }
      }
      __syncthreads();
    }
    // ---- Wef partial: newE += Z[16q x 512] @ WefT-slice.  wave w owns k_local
    // {2w, 2w+1}; accumulators persist across the whole k-loop.
    #pragma unroll
    for (int i2 = 0; i2 < 2; ++i2) {
      const int kk = wave * 2 + i2;
      #pragma unroll
      for (int nt = 0; nt < 2; ++nt) {
        v16bf a = load_a_frag(Zs, 16 * DE, kk * 32, lane);   // A[q][kl*32+e], ld=512
        v16bf bb = load_b_frag(WefT, NN * DE, nt * 16 + (lane & 15),
                               (kt * 16 + kk) * 32, lane);
        cW[nt] = wmma_bf16(a, bb, cW[nt]);
      }
    }
  }

  // write softmax partials
  {
    const size_t base = (size_t)(wg * SPLIT + split) * 2048;
    #pragma unroll
    for (int j = 0; j < 8; ++j) {
      int p = j * 256 + tid;
      partM[base + p] = sm[j]; partS[base + p] = ss[j]; partA[base + p] = sa[j];
    }
  }
  // deterministic cross-wave reduction of Wef partials via LDS staging
  #pragma unroll
  for (int nt = 0; nt < 2; ++nt) {
    #pragma unroll
    for (int r = 0; r < 8; ++r) {
      int q = r + 8 * (lane >> 4);
      int e = nt * 16 + (lane & 15);
      newEpart[(wave * 16 + q) * DE + e] = cW[nt][r];
    }
  }
  __syncthreads();
  for (int i = tid; i < 512; i += 256) {
    float s = 0.f;
    #pragma unroll
    for (int w = 0; w < 8; ++w) s += newEpart[(w * 16 + (i >> 5)) * DE + (i & 31)];
    partE[(size_t)(wg * SPLIT + split) * 512 + i] = s;
  }
}

// ---------------------------------------------------------------------------
// finalize: merge softmax splits, newX = wV @ Wxo (+bxo,*mask), newE = sum+bef
// ---------------------------------------------------------------------------
__global__ __launch_bounds__(256) void finalize(
    const float* __restrict__ partM, const float* __restrict__ partS,
    const float* __restrict__ partA, const float* __restrict__ partE,
    const bf16_t* __restrict__ WxoT, const float* __restrict__ bxo,
    const float* __restrict__ bef, const float* __restrict__ node_mask,
    float* __restrict__ outX, float* __restrict__ outE) {
  __shared__ __align__(16) bf16_t wVbf[16 * DX];
  __shared__ float maskQs[16];
  const int tid = threadIdx.x, lane = tid & 31, wave = tid >> 5;
  const int wg = blockIdx.x, b = wg >> 5, qt = wg & 31, q0 = qt * 16;
  if (tid < 16) maskQs[tid] = node_mask[b * NN + q0 + tid];
  #pragma unroll
  for (int j = 0; j < 8; ++j) {
    const int p = j * 256 + tid, q = p >> 7, d = p & 127;
    float M = neg_inf();
    #pragma unroll
    for (int s = 0; s < SPLIT; ++s)
      M = fmaxf(M, partM[(size_t)(wg * SPLIT + s) * 2048 + p]);
    float S = 0.f, A = 0.f;
    if (M > neg_inf()) {
      #pragma unroll
      for (int s = 0; s < SPLIT; ++s) {
        size_t ix = (size_t)(wg * SPLIT + s) * 2048 + p;
        float mi = partM[ix];
        if (mi > neg_inf()) {
          float sc = __expf(mi - M);
          S += partS[ix] * sc; A += partA[ix] * sc;
        }
      }
    }
    wVbf[q * DX + d] = f2bf(S > 0.f ? A / S : 0.f);
  }
  for (int i = tid; i < 512; i += 256) {
    float s = bef[i & 31];
    #pragma unroll
    for (int sp = 0; sp < SPLIT; ++sp) s += partE[(size_t)(wg * SPLIT + sp) * 512 + i];
    outE[(size_t)(b * NN + q0 + (i >> 5)) * DE + (i & 31)] = s;
  }
  __syncthreads();
  // newX: 16x128 = wV(16x128) @ WxoT ; wave w owns N-tile w
  v8f c = vzero8();
  #pragma unroll
  for (int ks = 0; ks < 4; ++ks) {
    v16bf a = load_a_frag(wVbf, DX, ks * 32, lane);
    v16bf bb = load_b_frag(WxoT, DX, wave * 16 + (lane & 15), ks * 32, lane);
    c = wmma_bf16(a, bb, c);
  }
  const int d = wave * 16 + (lane & 15);
  #pragma unroll
  for (int r = 0; r < 8; ++r) {
    int q = r + 8 * (lane >> 4);
    outX[(size_t)(b * NN + q0 + q) * DX + d] = (c[r] + bxo[d]) * maskQs[q];
  }
}

// ---------------------------------------------------------------------------
extern "C" void kernel_launch(void* const* d_in, const int* in_sizes, int n_in,
                              void* d_out, int out_size, void* d_ws, size_t ws_size,
                              hipStream_t stream) {
  const float* X    = (const float*)d_in[0];
  const float* E    = (const float*)d_in[1];
  const float* mask = (const float*)d_in[2];
  const float* Wq = (const float*)d_in[3];  const float* bq = (const float*)d_in[4];
  const float* Wk = (const float*)d_in[5];  const float* bk = (const float*)d_in[6];
  const float* Wv = (const float*)d_in[7];  const float* bv = (const float*)d_in[8];
  const float* Wem = (const float*)d_in[9];  const float* bem = (const float*)d_in[10];
  const float* Wea = (const float*)d_in[11]; const float* bea = (const float*)d_in[12];
  const float* Weo = (const float*)d_in[13]; const float* beo = (const float*)d_in[14];
  const float* Wef = (const float*)d_in[15]; const float* bef = (const float*)d_in[16];
  const float* Wxo = (const float*)d_in[17]; const float* bxo = (const float*)d_in[18];

  char* w = (char*)d_ws;
  float*  Qf   = (float*)(w + 0);
  float*  Kf   = (float*)(w + 524288);
  float*  Vf   = (float*)(w + 1048576);
  bf16_t* Xbf  = (bf16_t*)(w + 1572864);
  bf16_t* WqT  = (bf16_t*)(w + 1835008);
  bf16_t* WkT  = (bf16_t*)(w + 1867776);
  bf16_t* WvT  = (bf16_t*)(w + 1900544);
  bf16_t* WxoT = (bf16_t*)(w + 1933312);
  bf16_t* WemT = (bf16_t*)(w + 1966080);
  bf16_t* WeaT = (bf16_t*)(w + 1974272);
  bf16_t* WeoT = (bf16_t*)(w + 1982464);
  bf16_t* WefT = (bf16_t*)(w + 1990656);
  float*  partM = (float*)(w + 3039232);
  float*  partS = (float*)(w + 5136384);
  float*  partA = (float*)(w + 7233536);
  float*  partE = (float*)(w + 9330688);

  cvt_bf16<<<256, 256, 0, stream>>>(X, Xbf, BS * NN * DX);
  transpose_bf16<<<64, 256, 0, stream>>>(Wq,  WqT,  DX, DX);
  transpose_bf16<<<64, 256, 0, stream>>>(Wk,  WkT,  DX, DX);
  transpose_bf16<<<64, 256, 0, stream>>>(Wv,  WvT,  DX, DX);
  transpose_bf16<<<64, 256, 0, stream>>>(Wxo, WxoT, DX, DX);
  transpose_bf16<<<16, 256, 0, stream>>>(Wem, WemT, DE, DX);
  transpose_bf16<<<16, 256, 0, stream>>>(Wea, WeaT, DE, DX);
  transpose_bf16<<<16, 256, 0, stream>>>(Weo, WeoT, DX, DE);
  transpose_bf16<<<1024, 256, 0, stream>>>(Wef, WefT, NN * DE, DE);

  qkv_gemm<<<dim3(64, 3), 128, 0, stream>>>(Xbf, WqT, WkT, WvT, bq, bk, bv, mask,
                                            Qf, Kf, Vf);

  fused_attn<<<dim3(64, SPLIT), 256, 0, stream>>>(E, mask, Qf, Kf, Vf,
                                                  WemT, WeaT, WeoT, WefT,
                                                  bem, bea, beo,
                                                  partM, partS, partA, partE);

  float* outX = (float*)d_out;
  float* outE = outX + BS * NN * DX;
  finalize<<<64, 256, 0, stream>>>(partM, partS, partA, partE,
                                   WxoT, bxo, bef, mask, outX, outE);
}
